// TreeEncoder_31817117729336
// MI455X (gfx1250) — matrix operand order repaired and checked by reading, not compiled, not run on hardware
//
#include <hip/hip_runtime.h>

// ---------------- problem constants ----------------
#define HID       128
#define FEAT      128
#define B_TREES   32
#define EPT       1092
#define E_TOTAL   34944      // = 32*1092, also READ_PAD index
#define N_LEVELS  6

typedef __bf16 bf16;
typedef __attribute__((ext_vector_type(16))) bf16  v16bf;
typedef __attribute__((ext_vector_type(8)))  bf16  v8bf;
typedef __attribute__((ext_vector_type(8)))  float v8f;

// f32 -> bf16 round-to-nearest-even
static __device__ __forceinline__ bf16 f2bf(float f) {
    union { float f; unsigned u; } a; a.f = f;
    unsigned r = a.u + 0x7FFFu + ((a.u >> 16) & 1u);
    union { unsigned short s; bf16 b; } o; o.s = (unsigned short)(r >> 16);
    return o.b;
}

static __device__ __forceinline__ float sigmoidf(float x) {
    return 1.0f / (1.0f + __expf(-x));
}

// A fragment (16x32 bf16 tile) from LDS row-major [rows][ldk]
// lane<16: K = {0..7, 16..23}; lane>=16: K = {8..15, 24..31} (relative to ktile*32)
static __device__ __forceinline__ v16bf load_a_frag(const bf16* A, int ldk,
                                                    int m, int ktile, int lane) {
    int k0 = ktile * 32 + ((lane < 16) ? 0 : 8);
    const bf16* p = A + m * ldk + k0;
    union { v16bf v; v8bf h[2]; } u;
    u.h[0] = *(const v8bf*)(p);
    u.h[1] = *(const v8bf*)(p + 16);
    return u.v;
}

// B fragment from pre-packed global weights: packed[ntile][ktile][lane][16]
static __device__ __forceinline__ v16bf load_b_frag(const bf16* Wp, int KT,
                                                    int ntile, int kt, int lane) {
    return *(const v16bf*)(Wp + ((((size_t)ntile * KT + kt) * 32 + lane) << 4));
}

#define WMMA_BF16(acc, a, b) \
    __builtin_amdgcn_wmma_f32_16x16x32_bf16(false, (a), false, (b), (short)0, (acc), false, false)

// ---------------- weight packing ----------------
// dst element (nt, kt, lane, i) = bf16( W[kt*32 + (lane<16?0:8) + (i<8 ? i : 8+i)][nt*16 + lane%16] )
__global__ __launch_bounds__(256) void pack_weights(
    const float* __restrict__ Wz, const float* __restrict__ Wr,
    const float* __restrict__ Ur, const float* __restrict__ Wh,
    const float* __restrict__ Wa, bf16* __restrict__ ws)
{
    int e = blockIdx.x * 256 + threadIdx.x;          // 0 .. 131071
    const float* W; int KT; bf16* dst; int local;
    if (e < 32768)      { W = Wz; KT = 8; dst = ws;          local = e; }
    else if (e < 49152) { W = Wr; KT = 4; dst = ws + 32768;  local = e - 32768; }
    else if (e < 65536) { W = Ur; KT = 4; dst = ws + 49152;  local = e - 49152; }
    else if (e < 98304) { W = Wh; KT = 8; dst = ws + 65536;  local = e - 65536; }
    else                { W = Wa; KT = 8; dst = ws + 98304;  local = e - 98304; }
    int i    = local & 15;
    int lane = (local >> 4) & 31;
    int kt   = (local >> 9) % KT;
    int nt   = local / (512 * KT);
    int k = kt * 32 + ((lane < 16) ? 0 : 8) + ((i < 8) ? i : 8 + i);
    int n = nt * 16 + (lane & 15);
    dst[local] = f2bf(W[k * HID + n]);
}

// ---------------- per-level Tree-GRU kernel ----------------
// One block = 16 nodes, 8 waves; wave w owns output columns [16w, 16w+16).
__global__ __launch_bounds__(256) void level_kernel(
    float* __restrict__ Hbuf,
    const int* __restrict__ src_wid, const int* __restrict__ nei_idx,
    const int* __restrict__ out_idx, const float* __restrict__ emb,
    const float* __restrict__ bz, const float* __restrict__ bUr,
    const float* __restrict__ bh, const bf16* __restrict__ ws, int off)
{
    __shared__ bf16  xcat[16][256];     // [x | s] then [x | sum_gated]
    __shared__ bf16  hnei_bf[64][128];
    __shared__ float hnei_f[64][128];
    __shared__ float s_f[16][128];
    __shared__ float zbuf[16][128];
    __shared__ float xwr[16][128];

    const int tid  = threadIdx.x;
    const int lane = tid & 31;
    const int wave = tid >> 5;                  // N-tile
    const int base = off + blockIdx.x * 16;
    const int mA   = lane & 15;                 // A-fragment row
    const int cN   = wave * 16 + (lane & 15);   // output column

    const bf16* WzP = ws;
    const bf16* WrP = ws + 32768;
    const bf16* UrP = ws + 49152;
    const bf16* WhP = ws + 65536;

    // ---- Phase A: gather x and h_nei into LDS ----
    for (int e = tid; e < 16 * 128; e += 256) {
        int m = e >> 7, c = e & 127;
        int wid = src_wid[base + m];
        xcat[m][c] = f2bf(emb[(size_t)wid * FEAT + c]);
    }
    for (int e = tid; e < 64 * 128; e += 256) {
        int r = e >> 7, c = e & 127;
        int m = r >> 2, j = r & 3;
        int idx = nei_idx[(base + m) * 4 + j];
        float v = (idx >= E_TOTAL) ? 0.0f : Hbuf[(size_t)idx * HID + c];
        hnei_f[r][c]  = v;
        hnei_bf[r][c] = f2bf(v);
    }
    __syncthreads();
    for (int e = tid; e < 16 * 128; e += 256) {
        int m = e >> 7, c = e & 127;
        float s = hnei_f[m*4+0][c] + hnei_f[m*4+1][c]
                + hnei_f[m*4+2][c] + hnei_f[m*4+3][c];
        s_f[m][c] = s;
        xcat[m][128 + c] = f2bf(s);
    }
    __syncthreads();

    // ---- Phase B: z = sigmoid([x,s]@Wz + bz);  xwr = x@Wr ----
    {
        v8f acc = {};
        #pragma unroll
        for (int kt = 0; kt < 8; ++kt) {
            v16bf a = load_a_frag(&xcat[0][0], 256, mA, kt, lane);
            v16bf b = load_b_frag(WzP, 8, wave, kt, lane);
            acc = WMMA_BF16(acc, a, b);
        }
        float bias = bz[cN];
        #pragma unroll
        for (int k = 0; k < 8; ++k) {
            int m = k + ((lane < 16) ? 0 : 8);
            zbuf[m][cN] = sigmoidf(acc[k] + bias);
        }
    }
    {
        v8f acc = {};
        #pragma unroll
        for (int kt = 0; kt < 4; ++kt) {
            v16bf a = load_a_frag(&xcat[0][0], 256, mA, kt, lane);  // x part
            v16bf b = load_b_frag(WrP, 4, wave, kt, lane);
            acc = WMMA_BF16(acc, a, b);
        }
        #pragma unroll
        for (int k = 0; k < 8; ++k) {
            int m = k + ((lane < 16) ? 0 : 8);
            xwr[m][cN] = acc[k];
        }
    }
    __syncthreads();

    // ---- Phase C: r = sigmoid(xwr + h_nei@Ur + bUr); sum_gated = sum_j r*h ----
    {
        float bias = bUr[cN];
        #pragma unroll
        for (int t = 0; t < 4; ++t) {           // M-tiles of the 64-row h_nei GEMM
            v8f acc = {};
            #pragma unroll
            for (int kt = 0; kt < 4; ++kt) {
                v16bf a = load_a_frag(&hnei_bf[16 * t][0], 128, mA, kt, lane);
                v16bf b = load_b_frag(UrP, 4, wave, kt, lane);
                acc = WMMA_BF16(acc, a, b);
            }
            // this lane's 8 rows cover 2 nodes x 4 neighbors -> full local reduce
            float sg0 = 0.f, sg1 = 0.f;
            #pragma unroll
            for (int k = 0; k < 8; ++k) {
                int R = 16 * t + k + ((lane < 16) ? 0 : 8);
                int m = R >> 2;
                float r = sigmoidf(acc[k] + xwr[m][cN] + bias);
                float g = r * hnei_f[R][cN];
                if (k < 4) sg0 += g; else sg1 += g;
            }
            int m0 = (16 * t + ((lane < 16) ? 0 : 8)) >> 2;
            xcat[m0    ][128 + cN] = f2bf(sg0);
            xcat[m0 + 1][128 + cN] = f2bf(sg1);
        }
    }
    __syncthreads();

    // ---- Phase D: pre_h = tanh([x,sum_gated]@Wh + bh); scatter new_h ----
    {
        v8f acc = {};
        #pragma unroll
        for (int kt = 0; kt < 8; ++kt) {
            v16bf a = load_a_frag(&xcat[0][0], 256, mA, kt, lane);
            v16bf b = load_b_frag(WhP, 8, wave, kt, lane);
            acc = WMMA_BF16(acc, a, b);
        }
        float bias = bh[cN];
        #pragma unroll
        for (int k = 0; k < 8; ++k) {
            int m = k + ((lane < 16) ? 0 : 8);
            float pre = tanhf(acc[k] + bias);
            float z   = zbuf[m][cN];
            float nh  = (1.0f - z) * s_f[m][cN] + z * pre;
            int oi = out_idx[base + m];
            Hbuf[(size_t)oi * HID + cN] = nh;
        }
    }
}

// ---------------- root readout ----------------
__global__ __launch_bounds__(256) void root_kernel(
    const float* __restrict__ Hbuf, float* __restrict__ out_root,
    const int* __restrict__ root_wid, const int* __restrict__ root_nei,
    const float* __restrict__ emb, const float* __restrict__ ba,
    const bf16* __restrict__ WaP)
{
    __shared__ bf16 xcat[16][256];
    const int tid  = threadIdx.x;
    const int lane = tid & 31;
    const int wave = tid >> 5;
    const int base = blockIdx.x * 16;
    const int mA   = lane & 15;
    const int cN   = wave * 16 + (lane & 15);

    for (int e = tid; e < 16 * 128; e += 256) {
        int m = e >> 7, c = e & 127;
        int wid = root_wid[base + m];
        xcat[m][c] = f2bf(emb[(size_t)wid * FEAT + c]);
        float s = 0.f;
        for (int j = 0; j < 4; ++j) {
            int idx = root_nei[(base + m) * 4 + j];
            if (idx < E_TOTAL) s += Hbuf[(size_t)idx * HID + c];
        }
        xcat[m][128 + c] = f2bf(s);
    }
    __syncthreads();

    v8f acc = {};
    #pragma unroll
    for (int kt = 0; kt < 8; ++kt) {
        v16bf a = load_a_frag(&xcat[0][0], 256, mA, kt, lane);
        v16bf b = load_b_frag(WaP, 8, wave, kt, lane);
        acc = WMMA_BF16(acc, a, b);
    }
    float bias = ba[cN];
    #pragma unroll
    for (int k = 0; k < 8; ++k) {
        int m = k + ((lane < 16) ? 0 : 8);
        float v = acc[k] + bias;
        out_root[(size_t)(base + m) * HID + cN] = v > 0.f ? v : 0.f;
    }
}

// ---------------- launch ----------------
extern "C" void kernel_launch(void* const* d_in, const int* in_sizes, int n_in,
                              void* d_out, int out_size, void* d_ws, size_t ws_size,
                              hipStream_t stream) {
    const int*   src_wid  = (const int*)d_in[0];
    const int*   nei_idx  = (const int*)d_in[1];
    const int*   out_idx  = (const int*)d_in[2];
    const int*   root_wid = (const int*)d_in[3];
    const int*   root_nei = (const int*)d_in[4];
    const float* emb      = (const float*)d_in[5];
    const float* Wz       = (const float*)d_in[6];
    const float* bz       = (const float*)d_in[7];
    const float* Wr       = (const float*)d_in[8];
    const float* Ur       = (const float*)d_in[9];
    const float* bUr      = (const float*)d_in[10];
    const float* Wh       = (const float*)d_in[11];
    const float* bh       = (const float*)d_in[12];
    const float* Wa       = (const float*)d_in[13];
    const float* ba       = (const float*)d_in[14];

    float* out = (float*)d_out;          // rows [0, E_TOTAL) are Hbuf == msgs
    bf16*  ws  = (bf16*)d_ws;            // 131072 bf16 = 256 KB packed weights

    pack_weights<<<512, 256, 0, stream>>>(Wz, Wr, Ur, Wh, Wa, ws);

    const int sizes[N_LEVELS] = {23328, 7776, 2592, 864, 288, 96};
    int off = 0;
    for (int t = 0; t < N_LEVELS; ++t) {
        level_kernel<<<sizes[t] / 16, 256, 0, stream>>>(
            out, src_wid, nei_idx, out_idx, emb, bz, bUr, bh, ws, off);
        off += sizes[t];
    }

    root_kernel<<<2, 256, 0, stream>>>(
        out, out + (size_t)E_TOTAL * HID, root_wid, root_nei, emb, ba, ws + 98304);
}
